// MultiHeadAttn_80126909874682
// MI455X (gfx1250) — matrix operand order repaired and checked by reading, not compiled, not run on hardware
//
#include <hip/hip_runtime.h>
#include <hip/hip_bf16.h>

// ---------------------------------------------------------------------------
// MultiHeadAttn (post-LN, no mask) for MI455X / gfx1250, f16 WMMA pipeline.
//   SEQ=2048  BATCH=4  N_HEAD=16  D_HEAD=64  D_MODEL=1024
// ---------------------------------------------------------------------------

#define SEQ     2048
#define BATCH   4
#define NHEAD   16
#define DHEAD   64
#define DMODEL  1024
#define MROWS   (SEQ * BATCH)          // 8192 GEMM rows (row r = i*BATCH + b)
#define NQKV    (3 * DMODEL)           // 3072 fused qkv output columns

typedef _Float16 v16h __attribute__((ext_vector_type(16)));
typedef _Float16 v8h  __attribute__((ext_vector_type(8)));
typedef float    v8f  __attribute__((ext_vector_type(8)));

// ---------------- WMMA fragment helpers (ISA 7.12.2 layouts) ----------------

// A-matrix 16x32 f16, row-major source with stride lda (elements).
// lane<16: M=lane, K in {0..7, 16..23}; lane>=16: M=lane-16, K in {8..15, 24..31}
__device__ __forceinline__ v16h load_a_f16(const _Float16* base, int lda) {
  int lane = threadIdx.x & 31;
  int m    = lane & 15;
  int sel  = lane >> 4;
  const _Float16* p = base + m * lda + sel * 8;
  v8h c0 = *(const v8h*)(p);        // K = sel*8 .. sel*8+7
  v8h c1 = *(const v8h*)(p + 16);   // K = sel*8+16 .. sel*8+23
  return __builtin_shufflevector(c0, c1, 0,1,2,3,4,5,6,7,8,9,10,11,12,13,14,15);
}

// B-matrix 32x16 f16 from B^T stored row-major [N][K] with stride ldb.
// lane<16: N=lane, K in {0..7, 16..23}; lane>=16: N=lane-16, K in {8..15, 24..31}
__device__ __forceinline__ v16h load_b_f16(const _Float16* bt, int ldb) {
  int lane = threadIdx.x & 31;
  const _Float16* p = bt + (lane & 15) * ldb + (lane >> 4) * 8;
  v8h c0 = *(const v8h*)(p);
  v8h c1 = *(const v8h*)(p + 16);
  return __builtin_shufflevector(c0, c1, 0,1,2,3,4,5,6,7,8,9,10,11,12,13,14,15);
}

__device__ __forceinline__ v8f wmma_f16(v16h a, v16h b, v8f c) {
  return __builtin_amdgcn_wmma_f32_16x16x32_f16(false, a, false, b,
                                                (short)0, c, false, false);
}

// ---------------------------------------------------------------------------
// Kernel 1: f32 -> f16 conversions + weight transposes.
//   h16 [8192][1024]
//   wt  [3072][1024]  = (Wq | Wkv)^T   (row n holds column n of the weight)
//   wot [1024][1024]  = Wo^T
// ---------------------------------------------------------------------------
#define HT_ELEMS   (MROWS * DMODEL)      // 8388608
#define WT_ELEMS   (NQKV * DMODEL)       // 3145728
#define WOT_ELEMS  (DMODEL * DMODEL)     // 1048576

__global__ __launch_bounds__(256)
void mha_convert(const float* __restrict__ h, const float* __restrict__ wq,
                 const float* __restrict__ wkv, const float* __restrict__ wo,
                 _Float16* __restrict__ h16, _Float16* __restrict__ wt,
                 _Float16* __restrict__ wot) {
  int idx = blockIdx.x * 256 + threadIdx.x;
  if (idx < HT_ELEMS) {
    h16[idx] = (_Float16)h[idx];
  } else if (idx < HT_ELEMS + WT_ELEMS) {
    int e = idx - HT_ELEMS;
    int n = e >> 10, k = e & 1023;
    float v = (n < DMODEL) ? wq[k * DMODEL + n]
                           : wkv[k * (2 * DMODEL) + (n - DMODEL)];
    wt[e] = (_Float16)v;
  } else if (idx < HT_ELEMS + WT_ELEMS + WOT_ELEMS) {
    int e = idx - HT_ELEMS - WT_ELEMS;
    int n = e >> 10, k = e & 1023;
    wot[e] = (_Float16)wo[k * DMODEL + n];
  }
}

// ---------------------------------------------------------------------------
// Kernel 2: fused QKV projection GEMM.  Each wave: 64x64 output tile,
// K=1024 in 32-deep steps (16 wmma/step, 16 b128 loads/step ~ 1 load/wmma).
// Scatter: Q[b,n][i][d] (scaled 1/8), K[b,n][i][d], V^T[b,n][d][i].
// ---------------------------------------------------------------------------
__global__ __launch_bounds__(256)
void mha_qkv_gemm(const _Float16* __restrict__ h16, const _Float16* __restrict__ wt,
                  _Float16* __restrict__ qb, _Float16* __restrict__ kb,
                  _Float16* __restrict__ vtb) {
  const int tilesN = NQKV / 64;                       // 48
  int wid  = blockIdx.x * (blockDim.x >> 5) + (threadIdx.x >> 5);
  int tm   = wid / tilesN, tn = wid % tilesN;
  int row0 = tm * 64, col0 = tn * 64;

  v8f acc[16] = {};

  for (int kk = 0; kk < DMODEL; kk += 32) {
    __builtin_prefetch(h16 + (size_t)row0 * DMODEL + kk + 64, 0, 3);
    v16h a[4];
#pragma unroll
    for (int i = 0; i < 4; ++i)
      a[i] = load_a_f16(h16 + (size_t)(row0 + 16 * i) * DMODEL + kk, DMODEL);
#pragma unroll
    for (int j = 0; j < 4; ++j) {
      v16h b = load_b_f16(wt + (size_t)(col0 + 16 * j) * DMODEL + kk, DMODEL);
#pragma unroll
      for (int i = 0; i < 4; ++i)
        acc[i * 4 + j] = wmma_f16(a[i], b, acc[i * 4 + j]);
    }
  }

  int lane = threadIdx.x & 31;
  int cn   = lane & 15;
  int rsel = (lane >> 4) * 8;
#pragma unroll
  for (int i = 0; i < 4; ++i)
#pragma unroll
    for (int j = 0; j < 4; ++j)
#pragma unroll
      for (int v = 0; v < 8; ++v) {
        int row = row0 + i * 16 + rsel + v;
        int col = col0 + j * 16 + cn;
        float x = acc[i * 4 + j][v];
        int si = row >> 2;             // seq index
        int b  = row & 3;              // batch index
        if (col < DMODEL) {            // Q (fold in 1/sqrt(64))
          int n = col >> 6, d = col & 63;
          qb[((b * NHEAD + n) * SEQ + si) * DHEAD + d] = (_Float16)(x * 0.125f);
        } else if (col < 2 * DMODEL) { // K
          int c = col - DMODEL, n = c >> 6, d = c & 63;
          kb[((b * NHEAD + n) * SEQ + si) * DHEAD + d] = (_Float16)x;
        } else {                       // V, stored transposed [b,n][d][seq]
          int c = col - 2 * DMODEL, n = c >> 6, d = c & 63;
          vtb[((b * NHEAD + n) * DHEAD + d) * SEQ + si] = (_Float16)x;
        }
      }
}

// ---------------------------------------------------------------------------
// Kernel 3: flash attention.  Block = 4 waves; wave owns 16 query rows.
// Streams j in 64-wide chunks; online softmax; wave-private LDS slices
// (no barriers, only s_wait_dscnt at fragment<->row layout handoffs).
// Writes attn_vec f16 as [row=i*4+b][n*64+d] (A-layout for the out GEMM).
// ---------------------------------------------------------------------------
#define JCH 64

__global__ __launch_bounds__(128)
void mha_attn(const _Float16* __restrict__ qb, const _Float16* __restrict__ kb,
              const _Float16* __restrict__ vtb, _Float16* __restrict__ av) {
  __shared__ float     Sld[4][16][JCH];   // 16 KB
  __shared__ _Float16  Pld[4][16][JCH];   //  8 KB
  __shared__ float     alf[4][16];

  int w    = threadIdx.x >> 5;
  int lane = threadIdx.x & 31;
  int cn   = lane & 15;
  int rsel = (lane >> 4) * 8;

  int bh = blockIdx.y;                 // b*16 + n
  int b  = bh >> 4, n = bh & 15;
  int i0 = blockIdx.x * 64 + w * 16;   // first query row of this wave

  const _Float16* qp = qb  + (size_t)bh * SEQ * DHEAD;
  const _Float16* kp = kb  + (size_t)bh * SEQ * DHEAD;
  const _Float16* vp = vtb + (size_t)bh * DHEAD * SEQ;

  v16h aq0 = load_a_f16(qp + (size_t)i0 * DHEAD +  0, DHEAD);
  v16h aq1 = load_a_f16(qp + (size_t)i0 * DHEAD + 32, DHEAD);

  v8f o[4] = {};
  float mrow = -3.0e38f, lrow = 0.0f;  // valid in lanes 0..15

  for (int j0 = 0; j0 < SEQ; j0 += JCH) {
    // ---- S = Q K^T (scale already folded into Q): 8 wmma ----
#pragma unroll
    for (int jt = 0; jt < JCH / 16; ++jt) {
      v16h bk0 = load_b_f16(kp + (size_t)(j0 + jt * 16) * DHEAD +  0, DHEAD);
      v16h bk1 = load_b_f16(kp + (size_t)(j0 + jt * 16) * DHEAD + 32, DHEAD);
      v8f s = {};
      s = wmma_f16(aq0, bk0, s);
      s = wmma_f16(aq1, bk1, s);
#pragma unroll
      for (int v = 0; v < 8; ++v) Sld[w][rsel + v][jt * 16 + cn] = s[v];
    }
    asm volatile("s_wait_dscnt 0" ::: "memory");

    // ---- online softmax: lanes 0..15, one row each ----
    if (lane < 16) {
      float mx = mrow;
#pragma unroll 8
      for (int jj = 0; jj < JCH; ++jj) mx = fmaxf(mx, Sld[w][lane][jj]);
      float al = __expf(mrow - mx);
      float ls = 0.0f;
#pragma unroll 8
      for (int jj = 0; jj < JCH; ++jj) {
        float p = __expf(Sld[w][lane][jj] - mx);
        ls += p;
        Pld[w][lane][jj] = (_Float16)p;
      }
      lrow = lrow * al + ls;
      mrow = mx;
      alf[w][lane] = al;
    }
    asm volatile("s_wait_dscnt 0" ::: "memory");

    // ---- rescale O accumulators by per-row alpha (once per 64 keys) ----
    float al8[8];
#pragma unroll
    for (int v = 0; v < 8; ++v) al8[v] = alf[w][rsel + v];
#pragma unroll
    for (int f = 0; f < 4; ++f)
#pragma unroll
      for (int v = 0; v < 8; ++v) o[f][v] *= al8[v];

    // ---- O += P @ V  (V^T rows contiguous in j): 8 wmma ----
#pragma unroll
    for (int kc = 0; kc < JCH / 32; ++kc) {
      v16h ap = load_a_f16(&Pld[w][0][kc * 32], JCH);
#pragma unroll
      for (int dt = 0; dt < 4; ++dt) {
        v16h bv = load_b_f16(vp + (size_t)(dt * 16) * SEQ + j0 + kc * 32, SEQ);
        o[dt] = wmma_f16(ap, bv, o[dt]);
      }
    }
  }

  // ---- finalize: divide by l, write attn_vec ----
  if (lane < 16) alf[w][lane] = 1.0f / lrow;
  asm volatile("s_wait_dscnt 0" ::: "memory");
  float il8[8];
#pragma unroll
  for (int v = 0; v < 8; ++v) il8[v] = alf[w][rsel + v];
#pragma unroll
  for (int dt = 0; dt < 4; ++dt)
#pragma unroll
    for (int v = 0; v < 8; ++v) {
      int i = i0 + rsel + v;
      float x = o[dt][v] * il8[v];
      av[(size_t)(i * BATCH + b) * DMODEL + n * DHEAD + dt * 16 + cn] = (_Float16)x;
    }
}

// ---------------------------------------------------------------------------
// Kernel 4: out projection GEMM + residual.  y = av @ Wo + h  (f32).
// 64x64 wave tile like kernel 2.
// ---------------------------------------------------------------------------
__global__ __launch_bounds__(256)
void mha_outproj(const _Float16* __restrict__ av, const _Float16* __restrict__ wot,
                 const float* __restrict__ h, float* __restrict__ y) {
  const int tilesN = DMODEL / 64;                     // 16
  int wid  = blockIdx.x * (blockDim.x >> 5) + (threadIdx.x >> 5);
  int tm   = wid / tilesN, tn = wid % tilesN;
  int row0 = tm * 64, col0 = tn * 64;

  v8f acc[16] = {};

  for (int kk = 0; kk < DMODEL; kk += 32) {
    __builtin_prefetch(av + (size_t)row0 * DMODEL + kk + 64, 0, 3);
    v16h a[4];
#pragma unroll
    for (int i = 0; i < 4; ++i)
      a[i] = load_a_f16(av + (size_t)(row0 + 16 * i) * DMODEL + kk, DMODEL);
#pragma unroll
    for (int j = 0; j < 4; ++j) {
      v16h bfr = load_b_f16(wot + (size_t)(col0 + 16 * j) * DMODEL + kk, DMODEL);
#pragma unroll
      for (int i = 0; i < 4; ++i)
        acc[i * 4 + j] = wmma_f16(a[i], bfr, acc[i * 4 + j]);
    }
  }

  int lane = threadIdx.x & 31;
  int cn   = lane & 15;
  int rsel = (lane >> 4) * 8;
#pragma unroll
  for (int i = 0; i < 4; ++i)
#pragma unroll
    for (int j = 0; j < 4; ++j)
#pragma unroll
      for (int v = 0; v < 8; ++v) {
        int row = row0 + i * 16 + rsel + v;
        int col = col0 + j * 16 + cn;
        size_t off = (size_t)row * DMODEL + col;
        y[off] = acc[i * 4 + j][v] + h[off];
      }
}

// ---------------------------------------------------------------------------
// Kernel 5: LayerNorm over rows of y.
// ---------------------------------------------------------------------------
__global__ __launch_bounds__(256)
void mha_ln(const float* __restrict__ y, const float* __restrict__ g,
            const float* __restrict__ be, float* __restrict__ out) {
  __shared__ float red[256];
  __shared__ float mu_s, rs_s;
  int r = blockIdx.x, t = threadIdx.x;
  const float* yr = y + (size_t)r * DMODEL;

  float v[4], s = 0.0f;
#pragma unroll
  for (int u = 0; u < 4; ++u) { v[u] = yr[t + 256 * u]; s += v[u]; }
  red[t] = s; __syncthreads();
  for (int st = 128; st > 0; st >>= 1) {
    if (t < st) red[t] += red[t + st];
    __syncthreads();
  }
  if (t == 0) mu_s = red[0] * (1.0f / DMODEL);
  __syncthreads();
  float mu = mu_s;

  s = 0.0f;
#pragma unroll
  for (int u = 0; u < 4; ++u) { float d = v[u] - mu; s += d * d; }
  red[t] = s; __syncthreads();
  for (int st = 128; st > 0; st >>= 1) {
    if (t < st) red[t] += red[t + st];
    __syncthreads();
  }
  if (t == 0) rs_s = rsqrtf(red[0] * (1.0f / DMODEL) + 1e-5f);
  __syncthreads();
  float rs = rs_s;

#pragma unroll
  for (int u = 0; u < 4; ++u) {
    int c = t + 256 * u;
    out[(size_t)r * DMODEL + c] = (v[u] - mu) * rs * g[c] + be[c];
  }
}

// ---------------------------------------------------------------------------
// Host-side launcher.
// ---------------------------------------------------------------------------
extern "C" void kernel_launch(void* const* d_in, const int* in_sizes, int n_in,
                              void* d_out, int out_size, void* d_ws, size_t ws_size,
                              hipStream_t stream) {
  (void)in_sizes; (void)n_in; (void)out_size; (void)ws_size;

  const float* h     = (const float*)d_in[0];
  const float* Wq    = (const float*)d_in[1];
  const float* Wkv   = (const float*)d_in[2];
  const float* Wo    = (const float*)d_in[3];
  const float* gamma = (const float*)d_in[4];
  const float* beta  = (const float*)d_in[5];
  float*       out   = (float*)d_out;

  char* ws = (char*)d_ws;
  size_t off = 0;
  _Float16* h16 = (_Float16*)(ws + off); off += (size_t)HT_ELEMS  * 2;  // 16 MB
  _Float16* wt  = (_Float16*)(ws + off); off += (size_t)WT_ELEMS  * 2;  //  6 MB
  _Float16* wot = (_Float16*)(ws + off); off += (size_t)WOT_ELEMS * 2;  //  2 MB
  _Float16* qb  = (_Float16*)(ws + off); off += (size_t)HT_ELEMS  * 2;  // 16 MB
  _Float16* kb  = (_Float16*)(ws + off); off += (size_t)HT_ELEMS  * 2;  // 16 MB
  _Float16* vtb = (_Float16*)(ws + off); off += (size_t)HT_ELEMS  * 2;  // 16 MB
  _Float16* av  = (_Float16*)(ws + off); off += (size_t)HT_ELEMS  * 2;  // 16 MB
  float*    yb  = (float*)  (ws + off); off += (size_t)HT_ELEMS  * 4;   // 32 MB

  // 1) conversions + transposes
  {
    int total  = HT_ELEMS + WT_ELEMS + WOT_ELEMS;
    int blocks = (total + 255) / 256;
    mha_convert<<<blocks, 256, 0, stream>>>(h, Wq, Wkv, Wo, h16, wt, wot);
  }
  // 2) fused QKV GEMM: (8192/64)*(3072/64)=6144 waves, 8 waves/block
  mha_qkv_gemm<<<(MROWS / 64) * (NQKV / 64) / 8, 256, 0, stream>>>(h16, wt, qb, kb, vtb);
  // 3) flash attention: grid (i-tiles of 64, b*n combos)
  mha_attn<<<dim3(SEQ / 64, BATCH * NHEAD), 128, 0, stream>>>(qb, kb, vtb, av);
  // 4) out projection + residual: (8192/64)*(1024/64)=2048 waves, 8/block
  mha_outproj<<<(MROWS / 64) * (DMODEL / 64) / 8, 256, 0, stream>>>(av, wot, h, yb);
  // 5) layernorm, one block per row
  mha_ln<<<MROWS, 256, 0, stream>>>(yb, gamma, beta, out);
}
